// NeuronCircuit_28930899706061
// MI455X (gfx1250) — matrix operand order repaired and checked by reading, not compiled, not run on hardware
//
#include <hip/hip_runtime.h>

// ---------------------------------------------------------------------------
// Types
// ---------------------------------------------------------------------------
typedef __bf16 bf16_t;
typedef __attribute__((ext_vector_type(16))) __bf16 v16bf;
typedef __attribute__((ext_vector_type(8)))  __bf16 v8bf;
typedef __attribute__((ext_vector_type(8)))  float  v8f;

#define B_  2
#define S_  2048
#define D_  1024
#define H_  16
#define DH_ 64
#define RK_ 128
#define NC_ 16
#define ROWS_ (B_ * S_)        // 4096
#define YC_   (NC_ * RK_)      // 2048

// Async-to-LDS path (gfx1250): guarded so a missing builtin falls back cleanly.
#if defined(__gfx1250__) && __has_builtin(__builtin_amdgcn_global_load_async_to_lds_b128)
#define ASYNC_OK 1
// Builtin signature (from hipcc diagnostic): (v4i AS1*, v4i AS3*, imm, imm)
typedef int async_v4i __attribute__((vector_size(16)));
typedef __attribute__((address_space(1))) async_v4i* g_v4i_p;
typedef __attribute__((address_space(3))) async_v4i* l_v4i_p;
#else
#define ASYNC_OK 0
#endif

// ---------------------------------------------------------------------------
// bf16 helpers (bit-level, avoids relying on __bf16 conversion support)
// ---------------------------------------------------------------------------
__device__ __forceinline__ bf16_t f2bf(float f) {
  unsigned u = __builtin_bit_cast(unsigned, f);
  unsigned r = (u + 0x7FFFu + ((u >> 16) & 1u)) >> 16;
  return __builtin_bit_cast(bf16_t, (unsigned short)r);
}
__device__ __forceinline__ float bf2f(bf16_t b) {
  unsigned short s = __builtin_bit_cast(unsigned short, b);
  return __builtin_bit_cast(float, (unsigned)s << 16);
}

// ---------------------------------------------------------------------------
// WMMA fragment loads (wave32; layouts per CDNA5 ISA §7.12.2)
// ---------------------------------------------------------------------------
__device__ __forceinline__ v16bf load_a_frag(const bf16_t* base, int ld) {
  const int lane = threadIdx.x & 31;
  const int row  = lane & 15;
  const int kb   = (lane >> 4) << 3;             // 0 or 8
  const bf16_t* p = base + (long)row * ld + kb;
  v8bf lo = *(const v8bf*)(p);
  v8bf hi = *(const v8bf*)(p + 16);
  return __builtin_shufflevector(lo, hi, 0,1,2,3,4,5,6,7,8,9,10,11,12,13,14,15);
}

__device__ __forceinline__ v16bf load_b_frag(const bf16_t* bt, int ld) {
  const int lane = threadIdx.x & 31;
  const int n    = lane & 15;
  const int kb   = (lane >> 4) << 3;             // 0 or 8
  const bf16_t* p = bt + (long)n * ld + kb;
  v8bf lo = *(const v8bf*)(p);
  v8bf hi = *(const v8bf*)(p + 16);
  return __builtin_shufflevector(lo, hi, 0,1,2,3,4,5,6,7,8,9,10,11,12,13,14,15);
}

__device__ __forceinline__ v8f wmma_bf16(v16bf a, v16bf b, v8f c) {
  return __builtin_amdgcn_wmma_f32_16x16x32_bf16(
      false, a, false, b, (short)0, c, false, false);
}

__device__ __forceinline__ float rmax16(float v) {
  v = fmaxf(v, __shfl_xor(v, 1, 16));
  v = fmaxf(v, __shfl_xor(v, 2, 16));
  v = fmaxf(v, __shfl_xor(v, 4, 16));
  v = fmaxf(v, __shfl_xor(v, 8, 16));
  return v;
}
__device__ __forceinline__ float rsum16(float v) {
  v += __shfl_xor(v, 1, 16);
  v += __shfl_xor(v, 2, 16);
  v += __shfl_xor(v, 4, 16);
  v += __shfl_xor(v, 8, 16);
  return v;
}

// ---------------------------------------------------------------------------
// Conversion kernels (f32 -> bf16), plus compress_neurons transpose.
// ---------------------------------------------------------------------------
__global__ void conv_f2bf_kernel(const float* __restrict__ src,
                                 bf16_t* __restrict__ dst, int n) {
  int i = blockIdx.x * 256 + threadIdx.x;
  if (i < n) dst[i] = f2bf(src[i]);
}

// Ct[(n*RK + r), d] = C[n, d, r]   (C: [NC, D, RK])
__global__ void conv_ct_kernel(const float* __restrict__ C,
                               bf16_t* __restrict__ Ct) {
  int i = blockIdx.x * 256 + threadIdx.x;      // over YC_*D_
  int o = i >> 10;
  int d = i & 1023;
  int n = o >> 7, r = o & 127;
  Ct[i] = f2bf(C[(((size_t)n * D_) + d) * RK_ + r]);
}

// ---------------------------------------------------------------------------
// Router logits + softmax (3 groups of NC=16 = one N-tile each)
// ---------------------------------------------------------------------------
__global__ __launch_bounds__(32) void router_kernel(const bf16_t* __restrict__ X,
                                                    const bf16_t* __restrict__ R,
                                                    float* __restrict__ Wsoft) {
  const int rb = blockIdx.x * 16;
  const v8f zero = {0, 0, 0, 0, 0, 0, 0, 0};
  v8f acc[3] = {zero, zero, zero};
  const bf16_t* A = X + (size_t)rb * D_;
  for (int k = 0; k < D_; k += 32) {
    v16bf a = load_a_frag(A + k, D_);
#pragma unroll
    for (int g = 0; g < 3; ++g) {
      v16bf b = load_b_frag(R + (size_t)(g * 16) * D_ + k, D_);
      acc[g] = wmma_bf16(a, b, acc[g]);
    }
  }
  const int lane = threadIdx.x & 31;
  const int col = lane & 15, hi8 = (lane >> 4) << 3;
#pragma unroll
  for (int g = 0; g < 3; ++g) {
#pragma unroll
    for (int i = 0; i < 8; ++i) {
      float v = acc[g][i];
      float mr = rmax16(v);
      float e = __expf(v - mr);
      float s = rsum16(e);
      Wsoft[(size_t)(rb + i + hi8) * 48 + g * 16 + col] = e / s;
    }
  }
}

// ---------------------------------------------------------------------------
// Y = X[4096,1024] @ Ct^T -> [4096,2048] bf16.
// One wave per 32x64 tile; B tile double-buffered through LDS via async loads.
// ---------------------------------------------------------------------------
__global__ __launch_bounds__(32) void ygemm_kernel(const bf16_t* __restrict__ X,
                                                   const bf16_t* __restrict__ Ct,
                                                   bf16_t* __restrict__ Y) {
  const int rb = blockIdx.x * 32, cb = blockIdx.y * 64;
  const int lane = threadIdx.x & 31;
  const v8f zero = {0, 0, 0, 0, 0, 0, 0, 0};
  v8f acc[8] = {zero, zero, zero, zero, zero, zero, zero, zero};

#if ASYNC_OK
  __shared__ alignas(64) bf16_t bst[2][64 * 32];   // 2 x 4KB B tiles
  auto issue_stage = [&](int buf, int k) {
    char* lb = (char*)&bst[buf][0];
#pragma unroll
    for (int j = 0; j < 8; ++j) {
      const int o = j * 512 + lane * 16;           // byte offset in 4KB tile
      const int row = o >> 6;                      // 64B per tile row (32 bf16)
      const int colb = o & 63;
      const char* g =
          (const char*)Ct + (((size_t)(cb + row) * D_ + k) << 1) + colb;
      __builtin_amdgcn_global_load_async_to_lds_b128(
          (g_v4i_p)g, (l_v4i_p)(lb + o), 0, 0);
    }
  };
  issue_stage(0, 0);
#endif

  for (int k = 0; k < D_; k += 32) {
#if ASYNC_OK
    const int buf = (k >> 5) & 1;
    if (k + 32 < D_) {
      issue_stage(buf ^ 1, k + 32);
      asm volatile("s_wait_asynccnt 0x8" ::: "memory"); // prev stage complete
    } else {
      asm volatile("s_wait_asynccnt 0x0" ::: "memory");
    }
#endif
    v16bf a0 = load_a_frag(X + (size_t)rb * D_ + k, D_);
    v16bf a1 = load_a_frag(X + (size_t)(rb + 16) * D_ + k, D_);
#pragma unroll
    for (int j = 0; j < 4; ++j) {
#if ASYNC_OK
      v16bf b = load_b_frag(&bst[buf][j * 16 * 32], 32);
#else
      v16bf b = load_b_frag(Ct + (size_t)(cb + j * 16) * D_ + k, D_);
#endif
      acc[j]     = wmma_bf16(a0, b, acc[j]);
      acc[4 + j] = wmma_bf16(a1, b, acc[4 + j]);
    }
  }
  const int col = lane & 15, hi8 = (lane >> 4) << 3;
#pragma unroll
  for (int half = 0; half < 2; ++half)
#pragma unroll
    for (int j = 0; j < 4; ++j)
#pragma unroll
      for (int i = 0; i < 8; ++i)
        Y[(size_t)(rb + half * 16 + i + hi8) * YC_ + cb + j * 16 + col] =
            f2bf(acc[half * 4 + j][i]);
}

// ---------------------------------------------------------------------------
// h_g[row, r] = sum_n Wsoft[row, g*16+n] * Y[row, n*128 + r]
// ---------------------------------------------------------------------------
__global__ __launch_bounds__(256) void hmix_kernel(const float* __restrict__ Wsoft,
                                                   const bf16_t* __restrict__ Y,
                                                   bf16_t* __restrict__ Hout) {
  int tid = blockIdx.x * 256 + threadIdx.x;   // over 4096*128
  int row = tid >> 7, r = tid & 127;
  const bf16_t* yrow = Y + (size_t)row * YC_ + r;
  const float* wrow = Wsoft + (size_t)row * 48;
#pragma unroll
  for (int g = 0; g < 3; ++g) {
    float acc = 0.f;
#pragma unroll
    for (int n = 0; n < NC_; ++n) acc += wrow[g * 16 + n] * bf2f(yrow[n * RK_]);
    Hout[((size_t)g * ROWS_ + row) * RK_ + r] = f2bf(acc);
  }
}

// ---------------------------------------------------------------------------
// QKV up-projection (32x64 tiles): Q scaled & [bh,s,dh]; K [bh,s,dh]; V^T [bh,dh,s]
// ---------------------------------------------------------------------------
__global__ __launch_bounds__(32) void qkv_kernel(const bf16_t* __restrict__ Hin,
                                                 const bf16_t* __restrict__ W,
                                                 bf16_t* __restrict__ Qo,
                                                 bf16_t* __restrict__ Ko,
                                                 bf16_t* __restrict__ Vt) {
  const int t = blockIdx.z;
  const int rb = blockIdx.x * 32, cb = blockIdx.y * 64;
  const bf16_t* A  = Hin + (size_t)t * ROWS_ * RK_ + (size_t)rb * RK_;
  const bf16_t* Bt = W + (size_t)t * D_ * RK_;
  const v8f zero = {0, 0, 0, 0, 0, 0, 0, 0};
  v8f acc[8] = {zero, zero, zero, zero, zero, zero, zero, zero};
#pragma unroll
  for (int k = 0; k < RK_; k += 32) {
    v16bf a0 = load_a_frag(A + k, RK_);
    v16bf a1 = load_a_frag(A + (size_t)16 * RK_ + k, RK_);
#pragma unroll
    for (int j = 0; j < 4; ++j) {
      v16bf b = load_b_frag(Bt + (size_t)(cb + j * 16) * RK_ + k, RK_);
      acc[j]     = wmma_bf16(a0, b, acc[j]);
      acc[4 + j] = wmma_bf16(a1, b, acc[4 + j]);
    }
  }
  const int lane = threadIdx.x & 31;
  const int col = lane & 15, hi8 = (lane >> 4) << 3;
#pragma unroll
  for (int half = 0; half < 2; ++half)
#pragma unroll
    for (int j = 0; j < 4; ++j)
#pragma unroll
      for (int i = 0; i < 8; ++i) {
        const int row = rb + half * 16 + i + hi8;   // 0..4095
        const int bb = row >> 11, s = row & (S_ - 1);
        const int cg = cb + j * 16 + col;           // 0..1023
        const int head = cg >> 6, dh = cg & 63;
        const int bh = bb * H_ + head;
        const float v = acc[half * 4 + j][i];
        if (t == 0)
          Qo[((size_t)bh * S_ + s) * DH_ + dh] = f2bf(v * 0.125f);
        else if (t == 1)
          Ko[((size_t)bh * S_ + s) * DH_ + dh] = f2bf(v);
        else
          Vt[((size_t)bh * DH_ + dh) * S_ + s] = f2bf(v);
      }
}

// ---------------------------------------------------------------------------
// Causal flash attention, one wave per 16 query rows of one (b,h).
// ---------------------------------------------------------------------------
__global__ __launch_bounds__(32) void attn_kernel(const bf16_t* __restrict__ Q,
                                                  const bf16_t* __restrict__ K,
                                                  const bf16_t* __restrict__ Vt,
                                                  bf16_t* __restrict__ O) {
  __shared__ bf16_t plds[16 * 32];
  const int qb = blockIdx.x * 16;
  const int bh = blockIdx.y;                  // b*16 + h
  const int bb = bh >> 4, h = bh & 15;
  const bf16_t* Qb = Q + (size_t)bh * S_ * DH_;
  const bf16_t* Kb = K + (size_t)bh * S_ * DH_;
  const bf16_t* Vb = Vt + (size_t)bh * DH_ * S_;
  const int lane = threadIdx.x & 31;
  const int col = lane & 15, hi8 = (lane >> 4) << 3;

  const v16bf qa0 = load_a_frag(Qb + (size_t)qb * DH_ + 0, DH_);
  const v16bf qa1 = load_a_frag(Qb + (size_t)qb * DH_ + 32, DH_);

  const v8f zero = {0, 0, 0, 0, 0, 0, 0, 0};
  v8f acc0 = zero, acc1 = zero, acc2 = zero, acc3 = zero;
  float m[8], l[8];
#pragma unroll
  for (int i = 0; i < 8; ++i) { m[i] = -1e30f; l[i] = 0.f; }

  const int ntiles = (qb + 16 + 31) >> 5;     // causal early exit
  for (int kt = 0; kt < ntiles; ++kt) {
    const int kb = kt << 5;
    v8f s0 = zero, s1 = zero;
    s0 = wmma_bf16(qa0, load_b_frag(Kb + (size_t)kb * DH_ + 0, DH_), s0);
    s0 = wmma_bf16(qa1, load_b_frag(Kb + (size_t)kb * DH_ + 32, DH_), s0);
    s1 = wmma_bf16(qa0, load_b_frag(Kb + (size_t)(kb + 16) * DH_ + 0, DH_), s1);
    s1 = wmma_bf16(qa1, load_b_frag(Kb + (size_t)(kb + 16) * DH_ + 32, DH_), s1);
#pragma unroll
    for (int i = 0; i < 8; ++i) {
      const int q = qb + i + hi8;
      float a = ((kb + col) <= q) ? s0[i] : -1e30f;
      float c = ((kb + 16 + col) <= q) ? s1[i] : -1e30f;
      float mr = fmaxf(rmax16(fmaxf(a, c)), m[i]);
      const float sc = __expf(m[i] - mr);
      const float p0 = __expf(a - mr);
      const float p1 = __expf(c - mr);
      l[i] = l[i] * sc + rsum16(p0 + p1);
      m[i] = mr;
      acc0[i] *= sc; acc1[i] *= sc; acc2[i] *= sc; acc3[i] *= sc;
      plds[(i + hi8) * 32 + col] = f2bf(p0);
      plds[(i + hi8) * 32 + 16 + col] = f2bf(p1);
    }
    __syncthreads();                          // LDS relayout fence (single wave)
    v16bf pa = load_a_frag(plds, 32);
    acc0 = wmma_bf16(pa, load_b_frag(Vb + (size_t)0 * S_ + kb, S_), acc0);
    acc1 = wmma_bf16(pa, load_b_frag(Vb + (size_t)16 * S_ + kb, S_), acc1);
    acc2 = wmma_bf16(pa, load_b_frag(Vb + (size_t)32 * S_ + kb, S_), acc2);
    acc3 = wmma_bf16(pa, load_b_frag(Vb + (size_t)48 * S_ + kb, S_), acc3);
    __syncthreads();
  }
  bf16_t* Ob = O + ((size_t)(bb * S_ + qb)) * D_ + h * DH_;
#pragma unroll
  for (int i = 0; i < 8; ++i) {
    const float inv = 1.0f / l[i];
    bf16_t* orow = Ob + (size_t)(i + hi8) * D_;
    orow[ 0 + col] = f2bf(acc0[i] * inv);
    orow[16 + col] = f2bf(acc1[i] * inv);
    orow[32 + col] = f2bf(acc2[i] * inv);
    orow[48 + col] = f2bf(acc3[i] * inv);
  }
}

// ---------------------------------------------------------------------------
// out[4096,1024] = attn_out @ wO^T  (f32 output), 32x64 tiles
// ---------------------------------------------------------------------------
__global__ __launch_bounds__(32) void ogemm_kernel(const bf16_t* __restrict__ AO,
                                                   const bf16_t* __restrict__ WO,
                                                   float* __restrict__ out) {
  const int rb = blockIdx.x * 32, cb = blockIdx.y * 64;
  const v8f zero = {0, 0, 0, 0, 0, 0, 0, 0};
  v8f acc[8] = {zero, zero, zero, zero, zero, zero, zero, zero};
  const bf16_t* A = AO + (size_t)rb * D_;
  for (int k = 0; k < D_; k += 32) {
    __builtin_prefetch(A + k + 128, 0, 1);
    v16bf a0 = load_a_frag(A + k, D_);
    v16bf a1 = load_a_frag(A + (size_t)16 * D_ + k, D_);
#pragma unroll
    for (int j = 0; j < 4; ++j) {
      v16bf b = load_b_frag(WO + (size_t)(cb + j * 16) * D_ + k, D_);
      acc[j]     = wmma_bf16(a0, b, acc[j]);
      acc[4 + j] = wmma_bf16(a1, b, acc[4 + j]);
    }
  }
  const int lane = threadIdx.x & 31;
  const int col = lane & 15, hi8 = (lane >> 4) << 3;
#pragma unroll
  for (int half = 0; half < 2; ++half)
#pragma unroll
    for (int j = 0; j < 4; ++j)
#pragma unroll
      for (int i = 0; i < 8; ++i)
        out[(size_t)(rb + half * 16 + i + hi8) * D_ + cb + j * 16 + col] =
            acc[half * 4 + j][i];
}

// ---------------------------------------------------------------------------
// Host launcher
// ---------------------------------------------------------------------------
extern "C" void kernel_launch(void* const* d_in, const int* in_sizes, int n_in,
                              void* d_out, int out_size, void* d_ws, size_t ws_size,
                              hipStream_t stream) {
  (void)in_sizes; (void)n_in; (void)out_size; (void)ws_size;
  const float* x   = (const float*)d_in[0];
  // d_in[1] = mask (causal; implemented analytically)
  const float* cn  = (const float*)d_in[2];
  const float* rQ  = (const float*)d_in[3];
  const float* rK  = (const float*)d_in[4];
  const float* rV  = (const float*)d_in[5];
  const float* wQ  = (const float*)d_in[6];
  const float* wK  = (const float*)d_in[7];
  const float* wV  = (const float*)d_in[8];
  const float* wO  = (const float*)d_in[9];
  float* out = (float*)d_out;

  size_t off = 0;
  auto carve = [&](size_t bytes) {
    void* p = (char*)d_ws + off;
    off += (bytes + 255) & ~(size_t)255;
    return p;
  };
  bf16_t* Xbf   = (bf16_t*)carve((size_t)ROWS_ * D_ * 2);
  bf16_t* Ctbf  = (bf16_t*)carve((size_t)YC_ * D_ * 2);
  bf16_t* Rbf   = (bf16_t*)carve((size_t)48 * D_ * 2);
  bf16_t* Wqkv  = (bf16_t*)carve((size_t)3 * D_ * RK_ * 2);
  bf16_t* WObf  = (bf16_t*)carve((size_t)D_ * D_ * 2);
  float*  Wsoft = (float*)carve((size_t)ROWS_ * 48 * 4);
  bf16_t* Ybf   = (bf16_t*)carve((size_t)ROWS_ * YC_ * 2);
  bf16_t* Hbf   = (bf16_t*)carve((size_t)3 * ROWS_ * RK_ * 2);
  bf16_t* Qb    = (bf16_t*)carve((size_t)B_ * H_ * S_ * DH_ * 2);
  bf16_t* Kb    = (bf16_t*)carve((size_t)B_ * H_ * S_ * DH_ * 2);
  bf16_t* Vtb   = (bf16_t*)carve((size_t)B_ * H_ * S_ * DH_ * 2);
  bf16_t* AObf  = (bf16_t*)carve((size_t)ROWS_ * D_ * 2);

  conv_f2bf_kernel<<<(ROWS_ * D_) / 256, 256, 0, stream>>>(x, Xbf, ROWS_ * D_);
  conv_ct_kernel<<<(YC_ * D_) / 256, 256, 0, stream>>>(cn, Ctbf);
  conv_f2bf_kernel<<<64, 256, 0, stream>>>(rQ, Rbf + 0 * 16 * D_, NC_ * D_);
  conv_f2bf_kernel<<<64, 256, 0, stream>>>(rK, Rbf + 1 * 16 * D_, NC_ * D_);
  conv_f2bf_kernel<<<64, 256, 0, stream>>>(rV, Rbf + 2 * 16 * D_, NC_ * D_);
  conv_f2bf_kernel<<<512, 256, 0, stream>>>(wQ, Wqkv + 0 * D_ * RK_, D_ * RK_);
  conv_f2bf_kernel<<<512, 256, 0, stream>>>(wK, Wqkv + 1 * D_ * RK_, D_ * RK_);
  conv_f2bf_kernel<<<512, 256, 0, stream>>>(wV, Wqkv + 2 * D_ * RK_, D_ * RK_);
  conv_f2bf_kernel<<<4096, 256, 0, stream>>>(wO, WObf, D_ * D_);

  router_kernel<<<ROWS_ / 16, 32, 0, stream>>>(Xbf, Rbf, Wsoft);
  ygemm_kernel<<<dim3(ROWS_ / 32, YC_ / 64), 32, 0, stream>>>(Xbf, Ctbf, Ybf);
  hmix_kernel<<<(ROWS_ * RK_) / 256, 256, 0, stream>>>(Wsoft, Ybf, Hbf);
  qkv_kernel<<<dim3(ROWS_ / 32, D_ / 64, 3), 32, 0, stream>>>(Hbf, Wqkv, Qb, Kb, Vtb);
  attn_kernel<<<dim3(S_ / 16, B_ * H_), 32, 0, stream>>>(Qb, Kb, Vtb, AObf);
  ogemm_kernel<<<dim3(ROWS_ / 32, D_ / 64), 32, 0, stream>>>(AObf, WObf, out);
}
